// SlidingWindowAttention_89232240542340
// MI455X (gfx1250) — compile-verified, matrix operand order checked by reading, not run on hardware
//
#include <hip/hip_runtime.h>

// ---------------------------------------------------------------------------
// Types / helpers
// ---------------------------------------------------------------------------
typedef __attribute__((ext_vector_type(16))) __bf16 v16bf;
typedef __attribute__((ext_vector_type(8)))  float  v8f;

union BFrag {
  v16bf v;
  unsigned short u[16];
  uint4 q[2];
};

__device__ __forceinline__ unsigned short f2bf(float f) {
  unsigned int u = __float_as_uint(f);
  u += 0x7FFFu + ((u >> 16) & 1u);   // round-to-nearest-even
  return (unsigned short)(u >> 16);
}

__device__ __forceinline__ v16bf load_frag(const unsigned short* p0,
                                           const unsigned short* p1) {
  BFrag f;
  f.q[0] = *reinterpret_cast<const uint4*>(p0);
  f.q[1] = *reinterpret_cast<const uint4*>(p1);
  return f.v;
}

__device__ __forceinline__ v8f wmma_bf16(v16bf a, v16bf b, v8f c) {
  // D = A(16x32 bf16) x B(32x16 bf16) + C(16x16 f32)
  return __builtin_amdgcn_wmma_f32_16x16x32_bf16(false, a, false, b,
                                                 (short)0, c, false, false);
}

// Async copy 16 bytes global -> LDS (GLOBAL_LOAD_ASYNC_TO_LDS_B128, ASYNCcnt).
// VDST is a VGPR holding the LDS byte offset (= low 32 bits of a generic
// __shared__ pointer); VADDR is the 64-bit global address.
__device__ __forceinline__ void async_copy_b128(const unsigned short* lds_dst,
                                                const unsigned short* gsrc) {
  unsigned loff = (unsigned)(unsigned long long)(uintptr_t)lds_dst;
  unsigned long long gaddr = (unsigned long long)(uintptr_t)gsrc;
  asm volatile("global_load_async_to_lds_b128 %0, %1, off"
               :: "v"(loff), "v"(gaddr)
               : "memory");
}

__device__ __forceinline__ void wait_async0() {
  asm volatile("s_wait_asynccnt 0" ::: "memory");
}

// Problem constants
#define T_SEQ   8192
#define C_DIM   2048
#define HD      128
#define NHEAD   16
#define NKV     4
#define WIN     1024

// ---------------------------------------------------------------------------
// fp32 -> bf16 convert
// ---------------------------------------------------------------------------
__global__ void cvt_bf16_kernel(const float* __restrict__ in,
                                unsigned short* __restrict__ out, int n) {
  int i = blockIdx.x * blockDim.x + threadIdx.x;
  if (i < n) out[i] = f2bf(in[i]);
}

// W[C x N] fp32 -> WT[N x C] bf16
__global__ void transpose_bf16_kernel(const float* __restrict__ W,
                                      unsigned short* __restrict__ WT,
                                      int C, int N) {
  int i = blockIdx.x * blockDim.x + threadIdx.x;
  if (i < N * C) {
    int n = i / C;
    int c = i - n * C;
    WT[i] = f2bf(W[(size_t)c * N + n]);
  }
}

// ---------------------------------------------------------------------------
// bf16 WMMA GEMM: C[MxN] f32 = A[MxK] * BT[NxK]^T.
// Block = 128 threads (4 waves). Wave computes a 32x64 tile (2 M-tiles x 4
// N-tiles); block covers 128x64. The shared 64x32 B panel per K-step is
// staged into LDS with double-buffered async-to-LDS copies.
// ---------------------------------------------------------------------------
__global__ void __launch_bounds__(128)
gemm_bf16_kernel(const unsigned short* __restrict__ A,
                 const unsigned short* __restrict__ BT,
                 float* __restrict__ C,
                 int M, int N, int K) {
  __shared__ unsigned short tileB[2][64 * 32];   // 2 x 4KB

  const int lane = threadIdx.x & 31;
  const int wave = threadIdx.x >> 5;
  const int n16  = lane & 15;
  const int hi   = lane >> 4;
  const int mBase = blockIdx.y * 128 + wave * 32;
  const int nBase = blockIdx.x * 64;

  const unsigned short* aRow0 = A + (size_t)(mBase + n16) * K;
  const unsigned short* aRow1 = A + (size_t)(mBase + 16 + n16) * K;

  v8f acc[2][4];
#pragma unroll
  for (int mt = 0; mt < 2; ++mt)
#pragma unroll
    for (int t = 0; t < 4; ++t) acc[mt][t] = (v8f){};

  // staging map: 256 x 16B chunks; each of the 128 threads copies 2 chunks
  const int c0  = threadIdx.x * 2;
  const int sr0 = c0 >> 2,       sc0 = (c0 & 3) * 8;
  const int sr1 = (c0 + 1) >> 2, sc1 = ((c0 + 1) & 3) * 8;

  auto stage = [&](int buf, int k0) {
    async_copy_b128(&tileB[buf][sr0 * 32 + sc0],
                    BT + (size_t)(nBase + sr0) * K + k0 + sc0);
    async_copy_b128(&tileB[buf][sr1 * 32 + sc1],
                    BT + (size_t)(nBase + sr1) * K + k0 + sc1);
  };

  stage(0, 0);
  const int nSteps = K >> 5;
  for (int kk = 0; kk < nSteps; ++kk) {
    const int k0  = kk << 5;
    const int buf = kk & 1;
    wait_async0();          // this thread's copies for `buf` have landed
    __syncthreads();        // => everyone's copies for `buf` have landed
    if (kk + 1 < nSteps) stage(buf ^ 1, k0 + 32);   // overlap next panel

    __builtin_prefetch(aRow0 + k0 + 256, 0, 1);     // global_prefetch_b8
    v16bf a0 = load_frag(aRow0 + k0 + hi * 8, aRow0 + k0 + 16 + hi * 8);
    v16bf a1 = load_frag(aRow1 + k0 + hi * 8, aRow1 + k0 + 16 + hi * 8);

#pragma unroll
    for (int t = 0; t < 4; ++t) {
      const unsigned short* bp = &tileB[buf][(t * 16 + n16) * 32];
      v16bf bf = load_frag(bp + hi * 8, bp + 16 + hi * 8);   // ds_load_b128 x2
      acc[0][t] = wmma_bf16(a0, bf, acc[0][t]);
      acc[1][t] = wmma_bf16(a1, bf, acc[1][t]);
    }
    __syncthreads();        // done reading `buf` before it is re-staged
  }

  // C/D layout: lane holds column n16, rows hi*8 + r
#pragma unroll
  for (int mt = 0; mt < 2; ++mt) {
    float* cBase = C + (size_t)(mBase + mt * 16 + hi * 8) * N + nBase + n16;
#pragma unroll
    for (int r = 0; r < 8; ++r) {
      cBase[(size_t)r * N +  0] = acc[mt][0][r];
      cBase[(size_t)r * N + 16] = acc[mt][1][r];
      cBase[(size_t)r * N + 32] = acc[mt][2][r];
      cBase[(size_t)r * N + 48] = acc[mt][3][r];
    }
  }
}

// ---------------------------------------------------------------------------
// RoPE + RMS-norm + quant-dequant. One wave per (row, head) vector of 128.
// Waves [0,16T) -> Q, [16T,20T) -> K, [20T,24T) -> V (V stored transposed).
// ---------------------------------------------------------------------------
__global__ void qkv_post_kernel(const float* __restrict__ Qf,
                                const float* __restrict__ Kf,
                                const float* __restrict__ Vf,
                                const float* __restrict__ cosT,
                                const float* __restrict__ sinT,
                                unsigned short* __restrict__ Qb,
                                unsigned short* __restrict__ Kb,
                                unsigned short* __restrict__ Vt) {
  const int lane = threadIdx.x & 31;
  const int wid  = blockIdx.x * (blockDim.x >> 5) + (threadIdx.x >> 5);
  const int i0 = lane, i1 = lane + 32;

  if (wid < T_SEQ * NHEAD) {
    // ---- Q: rope + rms ----
    int h = wid / T_SEQ, t = wid - h * T_SEQ;
    const float* src = Qf + (size_t)t * C_DIM + h * HD;
    float c0 = cosT[t * 64 + i0], s0 = sinT[t * 64 + i0];
    float c1 = cosT[t * 64 + i1], s1 = sinT[t * 64 + i1];
    float a0 = src[i0], b0 = src[i0 + 64];
    float a1 = src[i1], b1 = src[i1 + 64];
    float v0 =  a0 * c0 + b0 * s0;
    float v1 =  a1 * c1 + b1 * s1;
    float v2 = -a0 * s0 + b0 * c0;
    float v3 = -a1 * s1 + b1 * c1;
    float ss = v0 * v0 + v1 * v1 + v2 * v2 + v3 * v3;
#pragma unroll
    for (int m = 16; m >= 1; m >>= 1) ss += __shfl_xor(ss, m, 32);
    float sc = rsqrtf(ss * (1.0f / 128.0f) + 1.1920929e-07f);
    unsigned short* dst = Qb + ((size_t)h * T_SEQ + t) * HD;
    dst[i0]      = f2bf(v0 * sc);
    dst[i1]      = f2bf(v1 * sc);
    dst[i0 + 64] = f2bf(v2 * sc);
    dst[i1 + 64] = f2bf(v3 * sc);
  } else if (wid < T_SEQ * (NHEAD + NKV)) {
    // ---- K: rope + rms + quant-dequant ----
    int r = wid - T_SEQ * NHEAD;
    int hk = r / T_SEQ, t = r - hk * T_SEQ;
    const float* src = Kf + (size_t)t * (NKV * HD) + hk * HD;
    float c0 = cosT[t * 64 + i0], s0 = sinT[t * 64 + i0];
    float c1 = cosT[t * 64 + i1], s1 = sinT[t * 64 + i1];
    float a0 = src[i0], b0 = src[i0 + 64];
    float a1 = src[i1], b1 = src[i1 + 64];
    float v0 =  a0 * c0 + b0 * s0;
    float v1 =  a1 * c1 + b1 * s1;
    float v2 = -a0 * s0 + b0 * c0;
    float v3 = -a1 * s1 + b1 * c1;
    float ss = v0 * v0 + v1 * v1 + v2 * v2 + v3 * v3;
#pragma unroll
    for (int m = 16; m >= 1; m >>= 1) ss += __shfl_xor(ss, m, 32);
    float sc = rsqrtf(ss * (1.0f / 128.0f) + 1.1920929e-07f);
    v0 *= sc; v1 *= sc; v2 *= sc; v3 *= sc;
    float am = fmaxf(fmaxf(fabsf(v0), fabsf(v1)), fmaxf(fabsf(v2), fabsf(v3)));
#pragma unroll
    for (int m = 16; m >= 1; m >>= 1) am = fmaxf(am, __shfl_xor(am, m, 32));
    float qs = fmaxf(am * (1.0f / 3.0f), 1e-8f);
    v0 = fminf(fmaxf(rintf(v0 / qs), -3.0f), 3.0f) * qs;
    v1 = fminf(fmaxf(rintf(v1 / qs), -3.0f), 3.0f) * qs;
    v2 = fminf(fmaxf(rintf(v2 / qs), -3.0f), 3.0f) * qs;
    v3 = fminf(fmaxf(rintf(v3 / qs), -3.0f), 3.0f) * qs;
    unsigned short* dst = Kb + ((size_t)hk * T_SEQ + t) * HD;
    dst[i0]      = f2bf(v0);
    dst[i1]      = f2bf(v1);
    dst[i0 + 64] = f2bf(v2);
    dst[i1 + 64] = f2bf(v3);
  } else if (wid < T_SEQ * (NHEAD + 2 * NKV)) {
    // ---- V: quant-dequant, store transposed [hk][d][t] ----
    int r = wid - T_SEQ * (NHEAD + NKV);
    int hk = r / T_SEQ, t = r - hk * T_SEQ;
    const float* src = Vf + (size_t)t * (NKV * HD) + hk * HD;
    float v0 = src[lane], v1 = src[lane + 32];
    float v2 = src[lane + 64], v3 = src[lane + 96];
    float am = fmaxf(fmaxf(fabsf(v0), fabsf(v1)), fmaxf(fabsf(v2), fabsf(v3)));
#pragma unroll
    for (int m = 16; m >= 1; m >>= 1) am = fmaxf(am, __shfl_xor(am, m, 32));
    float qs = fmaxf(am * (1.0f / 3.0f), 1e-8f);
    v0 = fminf(fmaxf(rintf(v0 / qs), -3.0f), 3.0f) * qs;
    v1 = fminf(fmaxf(rintf(v1 / qs), -3.0f), 3.0f) * qs;
    v2 = fminf(fmaxf(rintf(v2 / qs), -3.0f), 3.0f) * qs;
    v3 = fminf(fmaxf(rintf(v3 / qs), -3.0f), 3.0f) * qs;
    unsigned short* dst = Vt + (size_t)hk * HD * T_SEQ + t;
    dst[(size_t)(lane     ) * T_SEQ] = f2bf(v0);
    dst[(size_t)(lane + 32) * T_SEQ] = f2bf(v1);
    dst[(size_t)(lane + 64) * T_SEQ] = f2bf(v2);
    dst[(size_t)(lane + 96) * T_SEQ] = f2bf(v3);
  }
}

// ---------------------------------------------------------------------------
// Sliding-window flash attention. One wave per (head, 16-query tile).
// Computes S^T = K * Q^T so the score C-tile reinterprets directly as the
// B-fragment of P^T for the P*V WMMA (no cross-lane data movement).
// ---------------------------------------------------------------------------
__global__ void attn_kernel(const unsigned short* __restrict__ Qb,
                            const unsigned short* __restrict__ Kb,
                            const unsigned short* __restrict__ Vt,
                            unsigned short* __restrict__ Yb) {
  const int lane = threadIdx.x & 31;
  const int n16  = lane & 15;       // query column / key row / d row
  const int hi   = lane >> 4;
  const int gw   = blockIdx.x * (blockDim.x >> 5) + (threadIdx.x >> 5);
  const int h    = gw >> 9;                  // / (8192/16)
  const int t0   = (gw & 511) << 4;
  const int hk   = h >> 2;                   // GQA: 4 Q heads per KV head
  const int tq   = t0 + n16;                 // this lane's query index

  // Q^T B-fragments (d is the K-dim), loaded once
  v16bf qf[4];
  const unsigned short* qrow = Qb + ((size_t)h * T_SEQ + tq) * HD;
#pragma unroll
  for (int dc = 0; dc < 4; ++dc)
    qf[dc] = load_frag(qrow + dc * 32 + hi * 8, qrow + dc * 32 + 16 + hi * 8);

  v8f o[8];
#pragma unroll
  for (int i = 0; i < 8; ++i) o[i] = (v8f){};
  float mrun = -1e30f, lrun = 0.0f;
  const float scl = 0.08838834764831845f;    // 1/sqrt(128)

  const int jLo = (t0 >= WIN) ? ((t0 - (WIN - 1)) & ~31) : 0;
  const int jHi = (t0 + 15) & ~31;

  for (int j0 = jLo; j0 <= jHi; j0 += 32) {
    // --- S^T tiles: keys (rows) x queries (cols) ---
    v8f s0 = (v8f){}, s1 = (v8f){};
    const unsigned short* krow0 = Kb + ((size_t)hk * T_SEQ + j0 + n16) * HD;
    const unsigned short* krow1 = krow0 + (size_t)16 * HD;
#pragma unroll
    for (int dc = 0; dc < 4; ++dc) {
      v16bf ka0 = load_frag(krow0 + dc * 32 + hi * 8, krow0 + dc * 32 + 16 + hi * 8);
      s0 = wmma_bf16(ka0, qf[dc], s0);
      v16bf ka1 = load_frag(krow1 + dc * 32 + hi * 8, krow1 + dc * 32 + 16 + hi * 8);
      s1 = wmma_bf16(ka1, qf[dc], s1);
    }

    // --- mask + online softmax (stats are per-query = per-lane) ---
    float sv0[8], sv1[8];
    float mloc = -1e30f;
#pragma unroll
    for (int r = 0; r < 8; ++r) {
      int k0i = j0 + hi * 8 + r;
      int k1i = k0i + 16;
      float x0 = s0[r] * scl;
      float x1 = s1[r] * scl;
      x0 = (k0i <= tq && k0i > tq - WIN) ? x0 : -1e30f;
      x1 = (k1i <= tq && k1i > tq - WIN) ? x1 : -1e30f;
      sv0[r] = x0; sv1[r] = x1;
      mloc = fmaxf(mloc, fmaxf(x0, x1));
    }
    mloc = fmaxf(mloc, __shfl_xor(mloc, 16, 32));
    float mnew  = fmaxf(mrun, mloc);
    float alpha = __expf(mrun - mnew);

    BFrag pf;                      // P^T B-fragment: keys are the K-dim
    float rs = 0.0f;
#pragma unroll
    for (int r = 0; r < 8; ++r) {
      float p0 = (sv0[r] > -1e29f) ? __expf(sv0[r] - mnew) : 0.0f;
      float p1 = (sv1[r] > -1e29f) ? __expf(sv1[r] - mnew) : 0.0f;
      rs += p0 + p1;
      pf.u[r]     = f2bf(p0);
      pf.u[8 + r] = f2bf(p1);
    }
    rs += __shfl_xor(rs, 16, 32);
    lrun = lrun * alpha + rs;
    mrun = mnew;

    // --- O^T accumulate: A = V^T chunk (d rows x 32 keys), B = P^T ---
#pragma unroll
    for (int dt = 0; dt < 8; ++dt) {
#pragma unroll
      for (int r = 0; r < 8; ++r) o[dt][r] *= alpha;
      const unsigned short* vrow = Vt + ((size_t)hk * HD + dt * 16 + n16) * T_SEQ;
      v16bf va = load_frag(vrow + j0 + hi * 8, vrow + j0 + 16 + hi * 8);
      o[dt] = wmma_bf16(va, pf.v, o[dt]);
    }
  }

  // epilogue: O^T C-layout -> row-major Yb[t][h*128+d], divide by l
  float inv = 1.0f / lrun;
  unsigned short* dst = Yb + (size_t)tq * C_DIM + h * HD;
#pragma unroll
  for (int dt = 0; dt < 8; ++dt)
#pragma unroll
    for (int r = 0; r < 8; ++r)
      dst[dt * 16 + hi * 8 + r] = f2bf(o[dt][r] * inv);
}

// ---------------------------------------------------------------------------
// Launcher
// ---------------------------------------------------------------------------
extern "C" void kernel_launch(void* const* d_in, const int* in_sizes, int n_in,
                              void* d_out, int out_size, void* d_ws, size_t ws_size,
                              hipStream_t stream) {
  const float* x     = (const float*)d_in[0];
  const float* cosT  = (const float*)d_in[1];
  const float* sinT  = (const float*)d_in[2];
  const float* Wq    = (const float*)d_in[3];
  const float* Wk    = (const float*)d_in[4];
  const float* Wv    = (const float*)d_in[5];
  const float* Wproj = (const float*)d_in[6];
  float* out = (float*)d_out;

  char* ws = (char*)d_ws;
  size_t off = 0;
  unsigned short* xb     = (unsigned short*)(ws + off); off += (size_t)T_SEQ * C_DIM * 2;          // 32 MB
  unsigned short* WqT    = (unsigned short*)(ws + off); off += (size_t)C_DIM * C_DIM * 2;          // 8 MB
  unsigned short* WkT    = (unsigned short*)(ws + off); off += (size_t)(NKV * HD) * C_DIM * 2;     // 2 MB
  unsigned short* WvT    = (unsigned short*)(ws + off); off += (size_t)(NKV * HD) * C_DIM * 2;     // 2 MB
  unsigned short* WprojT = (unsigned short*)(ws + off); off += (size_t)C_DIM * C_DIM * 2;          // 8 MB
  float* Qf = (float*)(ws + off); size_t qfOff = off; off += (size_t)T_SEQ * C_DIM * 4;            // 64 MB
  float* Kf = (float*)(ws + off); off += (size_t)T_SEQ * (NKV * HD) * 4;                           // 16 MB
  float* Vf = (float*)(ws + off); off += (size_t)T_SEQ * (NKV * HD) * 4;                           // 16 MB
  unsigned short* Qb = (unsigned short*)(ws + off); off += (size_t)NHEAD * T_SEQ * HD * 2;         // 32 MB
  unsigned short* Kb = (unsigned short*)(ws + off); off += (size_t)NKV * T_SEQ * HD * 2;           // 8 MB
  unsigned short* Vt = (unsigned short*)(ws + off); off += (size_t)NKV * HD * T_SEQ * 2;           // 8 MB
  unsigned short* Yb = (unsigned short*)(ws + qfOff);          // reuse Qf region after postprocess

  // 1) convert x to bf16; transpose+convert weights
  {
    int n = T_SEQ * C_DIM;
    cvt_bf16_kernel<<<(n + 255) / 256, 256, 0, stream>>>(x, xb, n);
  }
  transpose_bf16_kernel<<<(C_DIM * C_DIM + 255) / 256, 256, 0, stream>>>(Wq, WqT, C_DIM, C_DIM);
  transpose_bf16_kernel<<<(NKV * HD * C_DIM + 255) / 256, 256, 0, stream>>>(Wk, WkT, C_DIM, NKV * HD);
  transpose_bf16_kernel<<<(NKV * HD * C_DIM + 255) / 256, 256, 0, stream>>>(Wv, WvT, C_DIM, NKV * HD);
  transpose_bf16_kernel<<<(C_DIM * C_DIM + 255) / 256, 256, 0, stream>>>(Wproj, WprojT, C_DIM, C_DIM);

  // 2) QKV projections (WMMA GEMMs, async-LDS staged B panels)
  gemm_bf16_kernel<<<dim3(C_DIM / 64, T_SEQ / 128), 128, 0, stream>>>(xb, WqT, Qf, T_SEQ, C_DIM, C_DIM);
  gemm_bf16_kernel<<<dim3((NKV * HD) / 64, T_SEQ / 128), 128, 0, stream>>>(xb, WkT, Kf, T_SEQ, NKV * HD, C_DIM);
  gemm_bf16_kernel<<<dim3((NKV * HD) / 64, T_SEQ / 128), 128, 0, stream>>>(xb, WvT, Vf, T_SEQ, NKV * HD, C_DIM);

  // 3) RoPE + RMS + quant-dequant (one wave per 128-vector)
  {
    int waves = T_SEQ * (NHEAD + 2 * NKV);     // 24 * T
    qkv_post_kernel<<<waves / 4, 128, 0, stream>>>(Qf, Kf, Vf, cosT, sinT, Qb, Kb, Vt);
  }

  // 4) sliding-window attention (one wave per head x 16-query tile)
  {
    int waves = NHEAD * (T_SEQ / 16);          // 8192
    attn_kernel<<<waves / 8, 256, 0, stream>>>(Qb, Kb, Vt, Yb);
  }

  // 5) output projection
  gemm_bf16_kernel<<<dim3(C_DIM / 64, T_SEQ / 128), 128, 0, stream>>>(Yb, WprojT, out, T_SEQ, C_DIM, C_DIM);
}
